// BNHNConv3D_8177617732126
// MI455X (gfx1250) — matrix operand order repaired and checked by reading, not compiled, not run on hardware
//
#include <hip/hip_runtime.h>
#include <stdint.h>

typedef __attribute__((ext_vector_type(16))) __bf16 v16bf;
typedef __attribute__((ext_vector_type(8)))  float  v8f;

#define EPS 1e-5f

__device__ __forceinline__ unsigned short f32_to_bf16(float f) {
    unsigned int u = __float_as_uint(f);
    unsigned int r = u + 0x7FFFu + ((u >> 16) & 1u);
    return (unsigned short)(r >> 16);
}

// ---------------- Kernel P1: per-block BN partial sums ----------------
// 256 blocks x 256 threads; block b covers voxels [b*1024, b*1024+1024).
__global__ __launch_bounds__(256)
void bn_partial_kernel(const float* __restrict__ x,
                       const unsigned char* __restrict__ mask,
                       float* __restrict__ ps1, float* __restrict__ ps2,
                       float* __restrict__ pn)
{
    __shared__ float red[256];
    __shared__ float redn[4];
    int tid = threadIdx.x;
    int c = tid & 63, vg = tid >> 6;
    int base = blockIdx.x * 1024;
    float s1 = 0.f, s2 = 0.f, nn = 0.f;
    for (int i = 0; i < 256; ++i) {
        int v = base + vg + i * 4;
        float m = mask[v] ? 1.f : 0.f;
        float xm = m * x[(size_t)v * 64 + c];
        s1 += xm; s2 += xm * xm;
        if (c == 0) nn += m;
    }
    red[tid] = s1;
    if (c == 0) redn[vg] = nn;
    __syncthreads();
    if (tid < 64)
        ps1[blockIdx.x * 64 + tid] = red[tid] + red[tid+64] + red[tid+128] + red[tid+192];
    if (tid == 0)
        pn[blockIdx.x] = redn[0] + redn[1] + redn[2] + redn[3];
    __syncthreads();
    red[tid] = s2;
    __syncthreads();
    if (tid < 64)
        ps2[blockIdx.x * 64 + tid] = red[tid] + red[tid+64] + red[tid+128] + red[tid+192];
}

// ---------------- Kernel P2: finalize BN stats + weight prep ----------------
__global__ __launch_bounds__(256)
void bn_finalize_kernel(const float* __restrict__ weight, const float* __restrict__ beta,
                        const float* __restrict__ gamma,
                        const float* __restrict__ a1p, const float* __restrict__ a2p,
                        const float* __restrict__ ps1, const float* __restrict__ ps2,
                        const float* __restrict__ pn,
                        float* __restrict__ g1o, float* __restrict__ a1bno,
                        float* __restrict__ Wg, float* __restrict__ Wb,
                        unsigned short* __restrict__ WT)
{
    __shared__ float s_g1[64], s_beta[64];
    int tid = threadIdx.x;
    float a1 = a1p[0], a2 = a2p[0];
    if (tid < 64) {
        float s1 = 0.f, s2 = 0.f, n = 0.f;
        for (int b = 0; b < 256; ++b) {
            s1 += ps1[b * 64 + tid];
            s2 += ps2[b * 64 + tid];
            n  += pn[b];
        }
        float mean = s1 / n;
        float ex2  = s2 / n;
        float var  = (ex2 - mean * mean) * n / (n - 1.f);
        float bstd = sqrtf(var + EPS);
        float bsp  = __powf(bstd, a2);
        float g1   = gamma[tid] / bsp;
        s_g1[tid]   = g1;
        s_beta[tid] = beta[tid];
        g1o[tid]    = g1;
        a1bno[tid]  = a1 * mean;
    }
    __syncthreads();
    // Collapsed broadcast-conv weights: Wg[t][co] = sum_ci g1[ci]*W[t][ci][co]
    for (int idx = tid; idx < 27 * 64; idx += 256) {
        int t = idx >> 6, co = idx & 63;
        float ag = 0.f, ab = 0.f;
        for (int ci = 0; ci < 64; ++ci) {
            float w = weight[((size_t)(t * 64 + ci)) * 64 + co];
            ag += s_g1[ci] * w;
            ab += s_beta[ci] * w;
        }
        Wg[idx] = ag; Wb[idx] = ab;
    }
    // Transposed bf16 weights WT[t][co][ci] (B-fragment friendly: ci contiguous)
    for (int idx = tid; idx < 27 * 64 * 64; idx += 256) {
        int t  = idx >> 12;
        int co = (idx >> 6) & 63;
        int ci = idx & 63;
        WT[idx] = f32_to_bf16(weight[((size_t)(t * 64 + ci)) * 64 + co]);
    }
}

// ---------------- Main kernel: fused stats + WMMA implicit-GEMM conv ----------------
// 16384 blocks x 128 threads (4 waves). Tile = 16 voxels along x, halo 18x3x3.
__global__ __launch_bounds__(128)
void bnhn_main_kernel(const float* __restrict__ x,
                      const unsigned char* __restrict__ mask,
                      const float* __restrict__ a1p, const float* __restrict__ a2p,
                      const float* __restrict__ g1v, const float* __restrict__ a1bnv,
                      const float* __restrict__ Wg, const float* __restrict__ Wb,
                      const unsigned short* __restrict__ WT,
                      float* __restrict__ out)
{
    __shared__ __align__(16) unsigned short s_in1[162 * 72]; // bf16 in1, padded rows
    __shared__ float s_xm[162 * 65];                          // f32 xm for stats
    __shared__ float s_m[162];
    __shared__ float s_invssp[16 * 64];
    __shared__ float s_mfac[16 * 64];
    __shared__ float s_Wg[27 * 64];
    __shared__ float s_Wb[27 * 64];
    __shared__ float s_g1[64];
    __shared__ float s_a1bn[64];

    int tid = threadIdx.x;
    int b = blockIdx.x;
    int x0 = (b & 3) << 4;
    int y  = (b >> 2) & 63;
    int z  = b >> 8;
    float a1 = a1p[0], a2 = a2p[0];

    for (int i = tid; i < 27 * 64; i += 128) { s_Wg[i] = Wg[i]; s_Wb[i] = Wb[i]; }
    if (tid < 64) { s_g1[tid] = g1v[tid]; s_a1bn[tid] = a1bnv[tid]; }
    for (int i = tid; i < 162; i += 128) {
        int hz = i / 54, r = i % 54, hy = r / 18, hx = r % 18;
        int gz = z - 1 + hz, gy = y - 1 + hy, gx = x0 - 1 + hx;
        float m = 0.f;
        if (gz >= 0 && gz < 64 && gy >= 0 && gy < 64 && gx >= 0 && gx < 64)
            m = mask[(gz * 64 + gy) * 64 + gx] ? 1.f : 0.f;
        s_m[i] = m;
    }
    __syncthreads();

    // Stage halo: xm (f32) and in1 = m*(x - a1*bn_mean)*g1 (bf16)
    for (int i = tid; i < 162 * 64; i += 128) {
        int hv = i >> 6, c = i & 63;
        float m = s_m[hv];
        float xm = 0.f, in1 = 0.f;
        if (m != 0.f) {
            int hz = hv / 54, r = hv % 54, hy = r / 18, hx = r % 18;
            int gz = z - 1 + hz, gy = y - 1 + hy, gx = x0 - 1 + hx;
            float xv = x[(size_t)((gz * 64 + gy) * 64 + gx) * 64 + c];
            xm = xv;
            in1 = (xv - s_a1bn[c]) * s_g1[c];
        }
        s_xm[hv * 65 + c] = xm;
        s_in1[hv * 72 + c] = f32_to_bf16(in1);
    }
    __syncthreads();

    // Per-voxel neighborhood stats -> 1/ssp and mean-factor
    for (int i = tid; i < 16 * 64; i += 128) {
        int v = i >> 6, c = i & 63;
        float cnt = 0.f, s1 = 0.f, s2 = 0.f;
        for (int dz = 0; dz < 3; ++dz)
            for (int dy = 0; dy < 3; ++dy)
                for (int dx = 0; dx < 3; ++dx) {
                    int hv = (dz * 3 + dy) * 18 + v + dx;
                    float m  = s_m[hv];
                    float xm = s_xm[hv * 65 + c];
                    cnt += m; s1 += xm; s2 += xm * xm;
                }
        float safe = fmaxf(cnt, 1.f);
        float mean = s1 / safe;
        float sqm  = s2 / safe;
        float vc   = fmaxf(cnt, 2.f);
        float var  = vc / (vc - 1.f) * fmaxf(sqm - mean * mean, 0.f);
        float sd   = sqrtf(var + EPS);
        float p    = __powf(sd, a2 - 1.f);       // = 1/ssp = stds^(a2-1)
        s_invssp[i] = p;
        s_mfac[i]   = mean * (a1 - 1.f) * p;     // means * -(1-a1) / ssp
    }
    __syncthreads();

    // WMMA implicit GEMM: M=16 voxels, N=16 co per wave, K=64 per tap (2x bf16 WMMA)
    int lane = tid & 31;
    int wave = tid >> 5;
    int row  = lane & 15;
    int hi   = lane >> 4;        // 0: lanes 0-15, 1: lanes 16-31
    int ls   = hi * 8;           // A-fragment K base within 32
    int kbB  = hi * 8;           // B-fragment K base within 32
    int co   = wave * 16 + row;
    v8f acc0 = {}, acc1 = {};

    for (int t = 0; t < 27; ++t) {
        int dz = t / 9, dy = (t / 3) % 3, dx = t % 3;
        int arow = ((dz * 3 + dy) * 18 + dx + row) * 72;
        const unsigned short* bp = WT + (((size_t)t * 64 + co) << 6);
        union { uint4 u[2]; v16bf v; } A, B;
        // K block 0 (ci 0..31)
        A.u[0] = *(const uint4*)(s_in1 + arow + ls);
        A.u[1] = *(const uint4*)(s_in1 + arow + ls + 16);
        B.u[0] = *(const uint4*)(bp + kbB);
        B.u[1] = *(const uint4*)(bp + kbB + 16);
        acc0 = __builtin_amdgcn_wmma_f32_16x16x32_bf16(false, A.v, false, B.v,
                                                       (short)0, acc0, false, false);
        // K block 1 (ci 32..63)
        A.u[0] = *(const uint4*)(s_in1 + arow + 32 + ls);
        A.u[1] = *(const uint4*)(s_in1 + arow + 32 + ls + 16);
        B.u[0] = *(const uint4*)(bp + 32 + kbB);
        B.u[1] = *(const uint4*)(bp + 32 + kbB + 16);
        acc1 = __builtin_amdgcn_wmma_f32_16x16x32_bf16(false, A.v, false, B.v,
                                                       (short)0, acc1, false, false);
    }

    // Hoist per-co collapsed weights into registers (indices constant after unroll)
    float wg[27], wb[27];
#pragma unroll
    for (int t = 0; t < 27; ++t) {
        wg[t] = s_Wg[t * 64 + co];
        wb[t] = s_Wb[t * 64 + co];
    }

    // Epilogue: C VGPR j holds M = j (lanes 0-15) / j+8 (lanes 16-31), N = lane&15
#pragma unroll
    for (int j = 0; j < 8; ++j) {
        int v = hi * 8 + j;
        float conv = acc0[j] + acc1[j];
        float g = 0.f, bt = 0.f;
        int t = 0;
#pragma unroll
        for (int dz = 0; dz < 3; ++dz)
#pragma unroll
            for (int dy = 0; dy < 3; ++dy)
#pragma unroll
                for (int dx = 0; dx < 3; ++dx, ++t) {
                    float m = s_m[(dz * 3 + dy) * 18 + v + dx];
                    g  = fmaf(m, wg[t], g);
                    bt = fmaf(m, wb[t], bt);
                }
        float mc  = s_m[72 + 1 + v]; // center voxel occupancy
        float val = (conv * s_invssp[v * 64 + co] + g * s_mfac[v * 64 + co] + bt) * mc;
        out[(((size_t)z * 64 + y) * 64 + (size_t)(x0 + v)) * 64 + co] = val;
    }
}

extern "C" void kernel_launch(void* const* d_in, const int* in_sizes, int n_in,
                              void* d_out, int out_size, void* d_ws, size_t ws_size,
                              hipStream_t stream)
{
    (void)in_sizes; (void)n_in; (void)out_size; (void)ws_size;
    const float* x            = (const float*)d_in[0];
    const unsigned char* mask = (const unsigned char*)d_in[1];
    const float* weight       = (const float*)d_in[2];
    const float* beta         = (const float*)d_in[3];
    const float* gamma        = (const float*)d_in[4];
    const float* mean_att     = (const float*)d_in[5];
    const float* std_att      = (const float*)d_in[6];
    float* out = (float*)d_out;

    char* ws = (char*)d_ws;
    float* ps1          = (float*)(ws);             // 256*64*4 = 65536 B
    float* ps2          = (float*)(ws + 65536);     // 65536 B
    float* pn           = (float*)(ws + 131072);    // 1024 B
    float* g1           = (float*)(ws + 132096);    // 256 B
    float* a1bn         = (float*)(ws + 132352);    // 256 B
    float* Wg           = (float*)(ws + 132608);    // 6912 B
    float* Wb           = (float*)(ws + 139520);    // 6912 B
    unsigned short* WT  = (unsigned short*)(ws + 146432); // 221184 B (16B-aligned)

    bn_partial_kernel<<<256, 256, 0, stream>>>(x, mask, ps1, ps2, pn);
    bn_finalize_kernel<<<1, 256, 0, stream>>>(weight, beta, gamma, mean_att, std_att,
                                              ps1, ps2, pn, g1, a1bn, Wg, Wb, WT);
    bnhn_main_kernel<<<16384, 128, 0, stream>>>(x, mask, mean_att, std_att,
                                                g1, a1bn, Wg, Wb, WT, out);
}
